// VecANDTree_43860206027289
// MI455X (gfx1250) — compile-verified, matrix operand order checked by reading, not run on hardware
//
#include <hip/hip_runtime.h>

typedef __attribute__((ext_vector_type(16))) _Float16 v16h;
typedef __attribute__((ext_vector_type(8)))  float    v8f;
typedef __attribute__((ext_vector_type(4)))  float    f32x4;

#define ROWS            65536
#define COLS            1024
#define ROWS_PER_WAVE   16
#define WAVES_PER_BLOCK 8

// One wave32 owns 16 rows. Accumulate rowsums with v_wmma_f32_16x16x32_f16
// against an all-ones B matrix; 32 K-steps cover the 1024-wide row.
// Per iteration each (lane, lane+16) pair consumes exactly one 128B cacheline
// of its row, so HBM lines are fetched once and fully used.
__global__ __launch_bounds__(256) void
VecANDTree_wmma_kernel(const float* __restrict__ x, float* __restrict__ out)
{
    const int lane    = threadIdx.x & 31;
    const int wave    = threadIdx.x >> 5;
    const int rowBase = (blockIdx.x * WAVES_PER_BLOCK + wave) * ROWS_PER_WAVE;
    const int m       = lane & 15;        // row within the 16-row tile
    const int half    = lane >> 4;        // 0: K {0..7,16..23}, 1: K {8..15,24..31}

    const float* rowPtr = x + (size_t)(rowBase + m) * COLS + half * 8;

    // B = all ones (any layout works: every column of D gets the row sums).
    v16h bOnes;
#pragma unroll
    for (int i = 0; i < 16; ++i) bOnes[i] = (_Float16)1.0f;

    v8f acc = {};   // f32 accumulator: exact for sums up to 1024

#pragma unroll 4
    for (int kk = 0; kk < COLS; kk += 32) {
        // 8 + 8 contiguous floats per lane (two 16B loads per block)
        f32x4 f0 = *(const f32x4*)(rowPtr + kk +  0);
        f32x4 f1 = *(const f32x4*)(rowPtr + kk +  4);
        f32x4 f2 = *(const f32x4*)(rowPtr + kk + 16);
        f32x4 f3 = *(const f32x4*)(rowPtr + kk + 20);

        // Pack into the 16-bit A-matrix 16x32 per-lane layout:
        // a[0..7] = first 8-K block, a[8..15] = second 8-K block.
        v16h a;
        a[0]  = (_Float16)f0.x; a[1]  = (_Float16)f0.y;
        a[2]  = (_Float16)f0.z; a[3]  = (_Float16)f0.w;
        a[4]  = (_Float16)f1.x; a[5]  = (_Float16)f1.y;
        a[6]  = (_Float16)f1.z; a[7]  = (_Float16)f1.w;
        a[8]  = (_Float16)f2.x; a[9]  = (_Float16)f2.y;
        a[10] = (_Float16)f2.z; a[11] = (_Float16)f2.w;
        a[12] = (_Float16)f3.x; a[13] = (_Float16)f3.y;
        a[14] = (_Float16)f3.z; a[15] = (_Float16)f3.w;

        // D = A x ones + C  -> every column of D holds the 16 partial rowsums
        acc = __builtin_amdgcn_wmma_f32_16x16x32_f16(
            /*neg_a=*/false, a, /*neg_b=*/false, bOnes,
            /*c_mod=*/(short)0, acc, /*reuse_a=*/false, /*reuse_b=*/false);
    }

    // D layout: lanes 0-15 hold rows 0-7 in acc[0..7]; lanes 16-31 hold rows 8-15.
    const int j = lane & 7;
    float s = acc[0];
    s = (j == 1) ? acc[1] : s;
    s = (j == 2) ? acc[2] : s;
    s = (j == 3) ? acc[3] : s;
    s = (j == 4) ? acc[4] : s;
    s = (j == 5) ? acc[5] : s;
    s = (j == 6) ? acc[6] : s;
    s = (j == 7) ? acc[7] : s;

    // Lanes 0-7 write rows 0-7, lanes 16-23 write rows 8-15 (distinct rows each).
    if ((lane & 8) == 0) {
        const int row = rowBase + half * 8 + j;
        out[row] = (s > 1023.5f) ? 1.0f : 0.0f;
    }
}

extern "C" void kernel_launch(void* const* d_in, const int* in_sizes, int n_in,
                              void* d_out, int out_size, void* d_ws, size_t ws_size,
                              hipStream_t stream)
{
    const float* x   = (const float*)d_in[0];
    float*       out = (float*)d_out;

    const int rowsPerBlock = ROWS_PER_WAVE * WAVES_PER_BLOCK;   // 128
    dim3 grid(ROWS / rowsPerBlock);                             // 512 blocks
    dim3 block(32 * WAVES_PER_BLOCK);                           // 256 threads = 8 waves

    VecANDTree_wmma_kernel<<<grid, block, 0, stream>>>(x, out);
}
